// SpectralMessage_24567212933541
// MI455X (gfx1250) — matrix operand order repaired and checked by reading, not compiled, not run on hardware
//
#include <hip/hip_runtime.h>
#include <hip/hip_bf16.h>
#include <cstdint>

#define NCAT 151
#define NRCH 51

typedef __attribute__((ext_vector_type(16))) __bf16 v16bf;
typedef __attribute__((ext_vector_type(8)))  float  v8f;

union Frag16 { v16bf v; uint4 q[2]; };

#if __has_builtin(__builtin_amdgcn_global_load_async_to_lds_b128)
#define HAS_ASYNC_LDS 1
typedef int v4i __attribute__((vector_size(16)));
typedef __attribute__((address_space(1))) v4i g_v4i;  // global int4
typedef __attribute__((address_space(3))) v4i l_v4i;  // LDS int4
#else
#define HAS_ASYNC_LDS 0
#endif

__device__ __forceinline__ void wait_async0() {
#if HAS_ASYNC_LDS
#if __has_builtin(__builtin_amdgcn_s_wait_asynccnt)
  __builtin_amdgcn_s_wait_asynccnt(0);
#else
  asm volatile("s_wait_asynccnt 0x0" ::: "memory");
#endif
#endif
}

__device__ __forceinline__ unsigned int f2bf(float f) {
  unsigned int u = __builtin_bit_cast(unsigned int, f);
  u += 0x7FFFu + ((u >> 16) & 1u);   // round-to-nearest-even
  return u >> 16;
}

// ---------------------------------------------------------------------------
// Transpose + convert: in[K][N] f32 (row-major) -> out[N][K] bf16 bits
// ---------------------------------------------------------------------------
__global__ __launch_bounds__(256)
void transpose_bf16(const float* __restrict__ in, unsigned short* __restrict__ out,
                    int K, int N) {
  int idx = blockIdx.x * 256 + threadIdx.x;
  if (idx >= K * N) return;
  int n = idx / K;
  int k = idx - n * K;
  out[(size_t)n * K + k] = (unsigned short)f2bf(in[(size_t)k * N + n]);
}

// ---------------------------------------------------------------------------
// bf16 WMMA GEMM: C[M][NDIM] = op(A[M][K]_f32 @ Bt[NDIM][K]_bf16)
// Block tile 128x128, 8 waves (4 along M x 2 along N), wave tile 32x64.
// Double-buffered LDS pipeline, one barrier per 32-deep K-step.
// B panel staged via GLOBAL_LOAD_ASYNC_TO_LDS_B128 when available (ASYNCcnt),
// A panel converted f32->bf16 in VALU (co-executes with the XDL/WMMA pipe).
// ---------------------------------------------------------------------------
constexpr int LDST = 40;  // LDS row stride in ushort (80B: 16B-aligned, conflict-free)

template<int NDIM, bool RELU, bool RESIDUAL>
__global__ __launch_bounds__(256)
void gemm_bf16_wmma(const float* __restrict__ A,
                    const unsigned short* __restrict__ Bt,
                    const float* __restrict__ Res,
                    float* __restrict__ C,
                    int M, int K) {
  __shared__ unsigned short As[2][128 * LDST];
  __shared__ unsigned short Bs[2][128 * LDST];

  const int tid  = threadIdx.x;
  const int lane = tid & 31;
  const int wid  = tid >> 5;
  const int wM   = wid & 3;    // 0..3 : 32 rows each
  const int wN   = wid >> 2;   // 0..1 : 64 cols each
  const int bm   = blockIdx.y * 128;
  const int bn   = blockIdx.x * 128;

  const v8f vzero = {0.f, 0.f, 0.f, 0.f, 0.f, 0.f, 0.f, 0.f};
  v8f acc[2][4];
#pragma unroll
  for (int i = 0; i < 2; ++i)
#pragma unroll
    for (int j = 0; j < 4; ++j) acc[i][j] = vzero;

  const int ra = tid >> 1;        // 0..127 (tile row)
  const int ca = (tid & 1) * 16;  // 0 or 16 (k sub-chunk)
  const int half = lane >> 4;     // lane group
  const int l16  = lane & 15;

  const float*          gA = A  + (size_t)(bm + ra) * K + ca;
  const unsigned short* gB = Bt + (size_t)(bn + ra) * K + ca;

  float4 fa[4];  // in-flight A global data (f32)

  auto loadA = [&](int kk) {
    const float4* g = (const float4*)(gA + kk);
    fa[0] = g[0]; fa[1] = g[1]; fa[2] = g[2]; fa[3] = g[3];
  };
  auto storeA = [&](int buf) {
    float v[16] = {fa[0].x, fa[0].y, fa[0].z, fa[0].w,
                   fa[1].x, fa[1].y, fa[1].z, fa[1].w,
                   fa[2].x, fa[2].y, fa[2].z, fa[2].w,
                   fa[3].x, fa[3].y, fa[3].z, fa[3].w};
    unsigned int* d = (unsigned int*)&As[buf][ra * LDST + ca];
#pragma unroll
    for (int i = 0; i < 8; ++i)
      d[i] = f2bf(v[2 * i]) | (f2bf(v[2 * i + 1]) << 16);
  };
  auto stageB = [&](int kk, int buf) {
#if HAS_ASYNC_LDS
    g_v4i* g0 = (g_v4i*)(uintptr_t)(gB + kk);
    l_v4i* l0 = (l_v4i*)(uintptr_t)&Bs[buf][ra * LDST + ca];
    __builtin_amdgcn_global_load_async_to_lds_b128(g0,     l0,     0, 0);
    __builtin_amdgcn_global_load_async_to_lds_b128(g0 + 1, l0 + 1, 0, 0);
#else
    const uint4* g = (const uint4*)(gB + kk);
    uint4* d = (uint4*)&Bs[buf][ra * LDST + ca];
    d[0] = g[0];
    d[1] = g[1];
#endif
  };

  // ---- prologue: fill buffer 0 ----
  loadA(0);
  storeA(0);
  stageB(0, 0);
  wait_async0();
  __syncthreads();

  int buf = 0;
  for (int kk = 0; kk < K; kk += 32) {
    const int nxt = kk + 32;
    if (nxt < K) loadA(nxt);  // global f32 loads in flight during compute

    // A frag: lane<16 row M=l16 holds K {0..7,16..23}; lane>=16 holds {8..15,24..31}
    Frag16 a[2], b[4];
#pragma unroll
    for (int mi = 0; mi < 2; ++mi) {
      int row = wM * 32 + mi * 16 + l16;
      a[mi].q[0] = *(const uint4*)&As[buf][row * LDST + half * 8];
      a[mi].q[1] = *(const uint4*)&As[buf][row * LDST + 16 + half * 8];
    }
    // B frag: N = l16, K = {8*half..+7, 16+8*half..+7} (matches A)
#pragma unroll
    for (int ni = 0; ni < 4; ++ni) {
      int col = wN * 64 + ni * 16 + l16;
      b[ni].q[0] = *(const uint4*)&Bs[buf][col * LDST + half * 8];
      b[ni].q[1] = *(const uint4*)&Bs[buf][col * LDST + 16 + half * 8];
    }

    if (nxt < K) {            // stage next tile into the other buffer
      stageB(nxt, buf ^ 1);
      storeA(buf ^ 1);
    }

#pragma unroll
    for (int mi = 0; mi < 2; ++mi)
#pragma unroll
      for (int ni = 0; ni < 4; ++ni)
        acc[mi][ni] = __builtin_amdgcn_wmma_f32_16x16x32_bf16(
            false, a[mi].v, false, b[ni].v, (short)0, acc[mi][ni], false, false);

    wait_async0();            // ASYNCcnt drained before the LDS barrier
    __syncthreads();
    buf ^= 1;
  }

  // C/D layout: element e -> M = 8*half + e, N = l16
#pragma unroll
  for (int mi = 0; mi < 2; ++mi) {
#pragma unroll
    for (int ni = 0; ni < 4; ++ni) {
      int col  = bn + wN * 64 + ni * 16 + l16;
      int row0 = bm + wM * 32 + mi * 16 + half * 8;
#pragma unroll
      for (int e = 0; e < 8; ++e) {
        float vv = acc[mi][ni][e];
        if (RELU) vv = vv > 0.f ? vv : 0.f;
        size_t idx = (size_t)(row0 + e) * NDIM + col;
        if (RESIDUAL) vv += Res[idx];
        C[idx] = vv;
      }
    }
  }
}

// ---------------------------------------------------------------------------
// conv1 fused with the freq_emb gather.
// x[b,r,i,j] = freq_emb[lab[i]*NCAT + lab[j], r]; y1 = conv3x3(x), 51 -> 10 ch.
// Grid: (32 batches, 10 row-strips), 256 threads.
// ---------------------------------------------------------------------------
__global__ __launch_bounds__(256)
void conv1_kernel(const float* __restrict__ freq_emb,
                  const int* __restrict__ obj_preds,
                  const float* __restrict__ w1,   // [10][51][3][3]
                  float* __restrict__ y1) {       // [32][10][100][100]
  __shared__ int   lab[100];
  __shared__ float w[NRCH * 9 * 10];  // [r][tap][o]
  const int b = blockIdx.x, strip = blockIdx.y, tid = threadIdx.x;

  for (int i = tid; i < 100; i += 256) lab[i] = obj_preds[b * 100 + i];
  for (int i = tid; i < NRCH * 9 * 10; i += 256) {
    int o = i / (NRCH * 9);
    int rt = i - o * (NRCH * 9);
    int r = rt / 9, tap = rt - r * 9;
    w[(r * 9 + tap) * 10 + o] = w1[i];
  }
  __syncthreads();

  for (int p = tid; p < 1000; p += 256) {
    int i = strip * 10 + p / 100;
    int j = p % 100;
    float acc[10];
#pragma unroll
    for (int o = 0; o < 10; ++o) acc[o] = 0.f;
    for (int di = -1; di <= 1; ++di) {
      int ii = i + di;
      if (ii < 0 || ii >= 100) continue;
      for (int dj = -1; dj <= 1; ++dj) {
        int jj = j + dj;
        if (jj < 0 || jj >= 100) continue;
        int tap = (di + 1) * 3 + (dj + 1);
        const float* fe = freq_emb + (size_t)(lab[ii] * NCAT + lab[jj]) * NRCH;
        for (int r = 0; r < NRCH; ++r) {
          float x = fe[r];
          const float* wp = &w[(r * 9 + tap) * 10];
#pragma unroll
          for (int o = 0; o < 10; ++o) acc[o] += wp[o] * x;
        }
      }
    }
#pragma unroll
    for (int o = 0; o < 10; ++o)
      y1[(((size_t)b * 10 + o) * 100 + i) * 100 + j] = acc[o];
  }
}

// ---------------------------------------------------------------------------
// conv2 (3x3, 10->5) + conv3 (1x1, 5->1) + tanh, fused. Grid (32, 10).
// ---------------------------------------------------------------------------
__global__ __launch_bounds__(256)
void conv23_kernel(const float* __restrict__ w2,   // [5][10][3][3]
                   const float* __restrict__ w3,   // [1][5][1][1]
                   const float* __restrict__ y1,   // [32][10][100][100]
                   float* __restrict__ adj) {      // [32][100][100]
  __shared__ float w2s[450];
  __shared__ float w3s[5];
  const int b = blockIdx.x, strip = blockIdx.y, tid = threadIdx.x;
  for (int i = tid; i < 450; i += 256) w2s[i] = w2[i];
  if (tid < 5) w3s[tid] = w3[tid];
  __syncthreads();

  for (int p = tid; p < 1000; p += 256) {
    int i = strip * 10 + p / 100;
    int j = p % 100;
    float acc[5];
#pragma unroll
    for (int o = 0; o < 5; ++o) acc[o] = 0.f;
    for (int di = -1; di <= 1; ++di) {
      int ii = i + di;
      if (ii < 0 || ii >= 100) continue;
      for (int dj = -1; dj <= 1; ++dj) {
        int jj = j + dj;
        if (jj < 0 || jj >= 100) continue;
        int tap = (di + 1) * 3 + (dj + 1);
#pragma unroll
        for (int ci = 0; ci < 10; ++ci) {
          float v = y1[(((size_t)b * 10 + ci) * 100 + ii) * 100 + jj];
#pragma unroll
          for (int o = 0; o < 5; ++o) acc[o] += w2s[(o * 10 + ci) * 9 + tap] * v;
        }
      }
    }
    float s = 0.f;
#pragma unroll
    for (int o = 0; o < 5; ++o) s += w3s[o] * acc[o];
    adj[((size_t)b * 100 + i) * 100 + j] = tanhf(s);
  }
}

// ---------------------------------------------------------------------------
// Batched matmul: ofl[b,n,:] = sum_m adj[b,n,m] * obj_u1[b,m,:]  (H=1024)
// Grid (n=100, b=32), 256 threads, float4 per thread, all f32.
// ---------------------------------------------------------------------------
__global__ __launch_bounds__(256)
void bmm_kernel(const float* __restrict__ adj,
                const float* __restrict__ obj_u1,
                float* __restrict__ ofl) {
  const int n = blockIdx.x, b = blockIdx.y, tid = threadIdx.x;
  __shared__ float arow[100];
  if (tid < 100) arow[tid] = adj[((size_t)b * 100 + n) * 100 + tid];
  __syncthreads();

  float4 acc = {0.f, 0.f, 0.f, 0.f};
  const float4* src = (const float4*)(obj_u1 + (size_t)b * 100 * 1024);
  for (int m = 0; m < 100; ++m) {
    float a = arow[m];
    float4 v = src[m * 256 + tid];
    acc.x += a * v.x; acc.y += a * v.y; acc.z += a * v.z; acc.w += a * v.w;
  }
  ((float4*)(ofl + ((size_t)b * 100 + n) * 1024))[tid] = acc;
}

// ---------------------------------------------------------------------------
extern "C" void kernel_launch(void* const* d_in, const int* in_sizes, int n_in,
                              void* d_out, int out_size, void* d_ws, size_t ws_size,
                              hipStream_t stream) {
  const float* E   = (const float*)d_in[0];   // [3200][4096]
  const float* Wc  = (const float*)d_in[1];   // [4096][1024]
  const float* Wu1 = (const float*)d_in[2];   // [1024][1024]
  const float* Wfc = (const float*)d_in[3];   // [1024][1024]
  const float* Wd  = (const float*)d_in[4];   // [1024][4096]
  const float* c1  = (const float*)d_in[5];   // [10][51][3][3]
  const float* c2  = (const float*)d_in[6];   // [5][10][3][3]
  const float* c3  = (const float*)d_in[7];   // [1][5][1][1]
  const float* fe  = (const float*)d_in[8];   // [151*151][51]
  const int*   lab = (const int*)d_in[9];     // [3200]
  (void)in_sizes; (void)n_in; (void)out_size; (void)ws_size;

  const int D = 4096, H = 1024, T = 3200;

  char* ws = (char*)d_ws;
  size_t off = 0;
  auto alloc = [&](size_t bytes) -> void* {
    void* p = ws + off;
    off = (off + bytes + 255) & ~(size_t)255;
    return p;
  };
  unsigned short* WcT  = (unsigned short*)alloc((size_t)D * H * 2);  // [H][D]
  unsigned short* Wu1T = (unsigned short*)alloc((size_t)H * H * 2);  // [H][H]
  unsigned short* WfcT = (unsigned short*)alloc((size_t)H * H * 2);  // [H][H]
  unsigned short* WdT  = (unsigned short*)alloc((size_t)H * D * 2);  // [D][H]
  float* comp = (float*)alloc((size_t)T * H * 4);                    // reused as u_m
  float* obju = (float*)alloc((size_t)T * H * 4);
  float* y1   = (float*)alloc((size_t)32 * 10 * 100 * 100 * 4);
  float* adj  = (float*)alloc((size_t)32 * 100 * 100 * 4);
  float* ofl  = (float*)alloc((size_t)T * H * 4);
  float* um   = comp;  // comp is dead after GEMM2

  // Weight transpose+convert (f32 [K][N] -> bf16 [N][K])
  transpose_bf16<<<(D * H + 255) / 256, 256, 0, stream>>>(Wc,  WcT,  D, H);
  transpose_bf16<<<(H * H + 255) / 256, 256, 0, stream>>>(Wu1, Wu1T, H, H);
  transpose_bf16<<<(H * H + 255) / 256, 256, 0, stream>>>(Wfc, WfcT, H, H);
  transpose_bf16<<<(H * D + 255) / 256, 256, 0, stream>>>(Wd,  WdT,  H, D);

  // Adjacency path (VALU): gather+conv1, conv2+conv3+tanh
  conv1_kernel<<<dim3(32, 10), 256, 0, stream>>>(fe, lab, c1, y1);
  conv23_kernel<<<dim3(32, 10), 256, 0, stream>>>(c2, c3, y1, adj);

  // GEMM1: comp = relu(E @ W_comp)          M=3200 N=1024 K=4096
  gemm_bf16_wmma<1024, true, false><<<dim3(1024 / 128, 3200 / 128), 256, 0, stream>>>(
      E, WcT, nullptr, comp, T, D);
  // GEMM2: obj_u1 = relu(comp @ W_ou1)      M=3200 N=1024 K=1024
  gemm_bf16_wmma<1024, true, false><<<dim3(1024 / 128, 3200 / 128), 256, 0, stream>>>(
      comp, Wu1T, nullptr, obju, T, H);
  // BMM: ofl = adj @ obj_u1 (per batch)
  bmm_kernel<<<dim3(100, 32), 256, 0, stream>>>(adj, obju, ofl);
  // GEMM3: u_m = relu(ofl @ W_ofc)          M=3200 N=1024 K=1024
  gemm_bf16_wmma<1024, true, false><<<dim3(1024 / 128, 3200 / 128), 256, 0, stream>>>(
      ofl, WfcT, nullptr, um, T, H);
  // GEMM4: out = relu(u_m @ W_decomp) + E   M=3200 N=4096 K=1024
  gemm_bf16_wmma<4096, true, true><<<dim3(4096 / 128, 3200 / 128), 256, 0, stream>>>(
      um, WdT, E, (float*)d_out, T, H);
}
